// xlstm_classifier_66374424592940
// MI455X (gfx1250) — matrix operand order repaired and checked by reading, not compiled, not run on hardware
//
#include <hip/hip_runtime.h>
#include <hip/hip_bf16.h>
#include <math.h>

// xLSTM classifier forward for MI455X (gfx1250), wave32, WMMA f16->f32 GEMMs.
#define BB   512
#define SS   64
#define FEATD 60
#define DM   256
#define NHD  4
#define INR  512
#define FFD  384
#define TT   (BB*SS)   // 32768 tokens

typedef __attribute__((ext_vector_type(16))) _Float16 v16h;
typedef __attribute__((ext_vector_type(8)))  float    v8f;

// ---------------------------------------------------------------------------
// WMMA GEMM:  C[M,N] = A[M,K] @ B  (+bias[N]) (+add[M,N])
//   WT==0: W is (K,N) row-major, ldw = row stride  -> B[k][n] = W[k*ldw+n]
//   WT==1: W is (N,K) row-major, ldw = row stride  -> B[k][n] = W[n*ldw+k]
// Each wave computes a 16x64 strip (4 accumulators, 4 v_wmma per K-step).
// REQUIRES: K % 32 == 0, M % 16 == 0, N % 64 == 0, 16B-aligned rows.
// No bounds checks -> no exec divergence, vector b128 loads.
// ---------------------------------------------------------------------------
template <int WT>
__global__ __launch_bounds__(256)
void wmma_gemm_kernel(const float* __restrict__ A, int lda,
                      const float* __restrict__ W, int ldw,
                      const float* __restrict__ bias,
                      const float* __restrict__ addp, int ldadd,
                      float* __restrict__ C, int ldc,
                      int M, int N, int K) {
  const int wave = threadIdx.x >> 5;
  const int lane = threadIdx.x & 31;
  const int stripsN = N >> 6;                     // 64-wide output strips
  const int strip = blockIdx.x * 8 + wave;
  const int total = (M >> 4) * stripsN;
  if (strip >= total) return;                     // wave-uniform
  const int m0 = (strip / stripsN) << 4;
  const int n0 = (strip % stripsN) << 6;
  const int row = m0 + (lane & 15);
  const int colB = n0 + (lane & 15);
  const int ka = (lane >> 4) << 4;                // 0 or 16  (A half-K select)
  const int kb = (lane >> 4) << 4;                // 0 or 16  (B half-K select)

  v8f acc[4] = {{}, {}, {}, {}};
  for (int k0 = 0; k0 < K; k0 += 32) {
    const float* ap = A + (size_t)row * lda + k0 + ka;
    float4 a0 = *(const float4*)(ap);
    float4 a1 = *(const float4*)(ap + 4);
    float4 a2 = *(const float4*)(ap + 8);
    float4 a3 = *(const float4*)(ap + 12);
    v16h af;
    af[0]  = (_Float16)a0.x; af[1]  = (_Float16)a0.y; af[2]  = (_Float16)a0.z; af[3]  = (_Float16)a0.w;
    af[4]  = (_Float16)a1.x; af[5]  = (_Float16)a1.y; af[6]  = (_Float16)a1.z; af[7]  = (_Float16)a1.w;
    af[8]  = (_Float16)a2.x; af[9]  = (_Float16)a2.y; af[10] = (_Float16)a2.z; af[11] = (_Float16)a2.w;
    af[12] = (_Float16)a3.x; af[13] = (_Float16)a3.y; af[14] = (_Float16)a3.z; af[15] = (_Float16)a3.w;
#pragma unroll
    for (int nt = 0; nt < 4; ++nt) {
      v16h bf;
      if (WT) {
        const float* wp = W + (size_t)(colB + nt * 16) * ldw + k0 + kb;
        float4 b0 = *(const float4*)(wp);
        float4 b1 = *(const float4*)(wp + 4);
        float4 b2 = *(const float4*)(wp + 8);
        float4 b3 = *(const float4*)(wp + 12);
        bf[0]  = (_Float16)b0.x; bf[1]  = (_Float16)b0.y; bf[2]  = (_Float16)b0.z; bf[3]  = (_Float16)b0.w;
        bf[4]  = (_Float16)b1.x; bf[5]  = (_Float16)b1.y; bf[6]  = (_Float16)b1.z; bf[7]  = (_Float16)b1.w;
        bf[8]  = (_Float16)b2.x; bf[9]  = (_Float16)b2.y; bf[10] = (_Float16)b2.z; bf[11] = (_Float16)b2.w;
        bf[12] = (_Float16)b3.x; bf[13] = (_Float16)b3.y; bf[14] = (_Float16)b3.z; bf[15] = (_Float16)b3.w;
      } else {
        const float* wp = W + (size_t)(k0 + kb) * ldw + colB + nt * 16;
#pragma unroll
        for (int i = 0; i < 16; ++i) bf[i] = (_Float16)wp[(size_t)i * ldw];  // coalesced across lanes
      }
      acc[nt] = __builtin_amdgcn_wmma_f32_16x16x32_f16(
          /*neg_a=*/false, af, /*neg_b=*/false, bf,
          /*c_mod=*/(short)0, acc[nt], /*reuse_a=*/false, /*reuse_b=*/false);
    }
  }
  const int rbase = m0 + ((lane >> 4) << 3);
#pragma unroll
  for (int nt = 0; nt < 4; ++nt) {
    const int cc = colB + nt * 16;
#pragma unroll
    for (int r = 0; r < 8; ++r) {
      int rr = rbase + r;
      float v = acc[nt][r];
      if (bias) v += bias[cc];
      if (addp) v += addp[(size_t)rr * ldadd + cc];
      C[(size_t)rr * ldc + cc] = v;
    }
  }
}

// -------- zero-pad input features 60 -> 64 so the GEMM needs no K guards ----
__global__ void pad_kernel(const float* __restrict__ x, float* __restrict__ out, long N) {
  long id = (long)blockIdx.x * blockDim.x + threadIdx.x;
  if (id >= N) return;                            // N = TT*64
  int k = (int)(id & 63);
  long t = id >> 6;
  out[id] = (k < FEATD) ? x[t * FEATD + k] : 0.f;
}

// ------------------------- LayerNorm (last dim = 256) -----------------------
__global__ __launch_bounds__(256)
void ln_kernel(const float* __restrict__ x, const float* __restrict__ w,
               float* __restrict__ out) {
  const int t = blockIdx.x, tid = threadIdx.x;   // blockDim == DM
  __shared__ float red[DM];
  const size_t base = (size_t)t * DM;
  float val = x[base + tid];
  red[tid] = val; __syncthreads();
  for (int off = DM >> 1; off > 0; off >>= 1) {
    if (tid < off) red[tid] += red[tid + off];
    __syncthreads();
  }
  float mean = red[0] * (1.f / DM); __syncthreads();
  float d = val - mean;
  red[tid] = d * d; __syncthreads();
  for (int off = DM >> 1; off > 0; off >>= 1) {
    if (tid < off) red[tid] += red[tid + off];
    __syncthreads();
  }
  float var = red[0] * (1.f / DM);
  out[base + tid] = d * rsqrtf(var + 1e-5f) * w[tid];
}

// ----------------- per-head group norm (scale only, opt. residual) ----------
__global__ void mhnorm_kernel(const float* __restrict__ x, const float* __restrict__ w,
                              const float* __restrict__ res, float* __restrict__ out,
                              int D) {
  const int gs = blockDim.x;                      // group size (64 or 128)
  const int groups = D / gs;
  const int t = blockIdx.x / groups;
  const int g = blockIdx.x % groups;
  const int tid = threadIdx.x;
  __shared__ float red[128];
  const size_t base = (size_t)t * D + g * gs;
  float val = x[base + tid];
  red[tid] = val; __syncthreads();
  for (int off = gs >> 1; off > 0; off >>= 1) {
    if (tid < off) red[tid] += red[tid + off];
    __syncthreads();
  }
  float mean = red[0] / gs; __syncthreads();
  float d = val - mean;
  red[tid] = d * d; __syncthreads();
  for (int off = gs >> 1; off > 0; off >>= 1) {
    if (tid < off) red[tid] += red[tid + off];
    __syncthreads();
  }
  float var = red[0] / gs;
  float y = d * rsqrtf(var + 1e-5f) * w[g * gs + tid];
  out[base + tid] = res ? (res[base + tid] + y) : y;
}

// ---------------- depthwise causal conv1d (K=4), optional SiLU --------------
__global__ void conv_silu_kernel(const float* __restrict__ x, int lda,
                                 const float* __restrict__ cw, const float* __restrict__ cb,
                                 float* __restrict__ out, int ldo, int C, long N, int doSilu) {
  long id = (long)blockIdx.x * blockDim.x + threadIdx.x;
  if (id >= N) return;
  int c = (int)(id % C);
  long ts = id / C;
  int s = (int)(ts % SS);
  long b = ts / SS;
  float acc = cb[c];
#pragma unroll
  for (int kk = 0; kk < 4; ++kk) {
    int sp = s - 3 + kk;
    if (sp >= 0) acc += x[(size_t)(b * SS + sp) * lda + c] * cw[c * 4 + kk];
  }
  if (doSilu) acc = acc / (1.f + expf(-acc));
  out[(size_t)(b * SS + s) * ldo + c] = acc;
}

// ----------------- block-diagonal 4x4 linear (128 blocks) -------------------
__global__ void blockdiag_kernel(const float* __restrict__ x, int lda,
                                 const float* __restrict__ w,
                                 float* __restrict__ out, long N) {  // N = TT*INR
  long id = (long)blockIdx.x * blockDim.x + threadIdx.x;
  if (id >= N) return;
  int f = (int)(id % INR);
  long t = id / INR;
  int nb = f >> 2, o = f & 3;
  const float* xr = x + (size_t)t * lda + (nb << 2);
  const float* wr = w + nb * 16 + o * 4;
  out[(size_t)t * INR + f] = xr[0]*wr[0] + xr[1]*wr[1] + xr[2]*wr[2] + xr[3]*wr[3];
}

// -------- mLSTM input/forget gates: (qkv concat) @ w + b  -> (B,NH,S) -------
__global__ __launch_bounds__(256)
void mlstm_gates_kernel(const float* __restrict__ q, const float* __restrict__ k,
                        const float* __restrict__ v,
                        const float* __restrict__ wi, const float* __restrict__ bi,
                        const float* __restrict__ wf, const float* __restrict__ bf,
                        float* __restrict__ ip, float* __restrict__ fp) {
  const int t = blockIdx.x;                 // token index b*SS+s
  const int b = t / SS, s = t % SS;
  const int tid = threadIdx.x;
  float acc[8] = {0,0,0,0,0,0,0,0};
  for (int f = tid; f < INR; f += 256) {
    float qv = q[(size_t)t * INR + f];
    float kv = k[(size_t)t * INR + f];
    float vv = v[(size_t)t * INR + f];
#pragma unroll
    for (int h = 0; h < NHD; ++h) {
      acc[h]     += qv * wi[f*NHD+h] + kv * wi[(INR+f)*NHD+h] + vv * wi[(2*INR+f)*NHD+h];
      acc[4 + h] += qv * wf[f*NHD+h] + kv * wf[(INR+f)*NHD+h] + vv * wf[(2*INR+f)*NHD+h];
    }
  }
  __shared__ float red[256];
  for (int g = 0; g < 8; ++g) {
    red[tid] = acc[g]; __syncthreads();
    for (int off = 128; off > 0; off >>= 1) {
      if (tid < off) red[tid] += red[tid + off];
      __syncthreads();
    }
    if (tid == 0) {
      int h = g & 3;
      float val = red[0] + ((g < 4) ? bi[h] : bf[h]);
      float* dst = (g < 4) ? ip : fp;
      dst[((size_t)b * NHD + h) * SS + s] = val;
    }
    __syncthreads();
  }
}

// ----------------- parallel stabilized mLSTM cell (dh=128) ------------------
__global__ __launch_bounds__(64)
void mlstm_cell_kernel(const float* __restrict__ q, const float* __restrict__ k,
                       const float* __restrict__ v,
                       const float* __restrict__ ip, const float* __restrict__ fp,
                       float* __restrict__ hout /* aliases q buffer */) {
  const int DHm = 128;
  const int b = blockIdx.x >> 2, hd = blockIdx.x & 3;
  const int s = threadIdx.x;                        // row owner, 0..63
  __shared__ float ksm[SS][DHm];                    // K tile, reused for V (32 KB)
  __shared__ float Psm[SS][SS + 1];                 // normalized weights (16.6 KB)
  __shared__ float cs[SS + 1], lfz[SS], ipz[SS];
  const size_t rowbase = ((size_t)(b * SS + s)) * INR + hd * DHm;
  for (int d = 0; d < DHm; ++d) ksm[s][d] = k[rowbase + d];
  {
    float x = fp[((size_t)b * NHD + hd) * SS + s];
    lfz[s] = (x >= 0.f) ? -log1pf(expf(-x)) : (x - log1pf(expf(x)));  // log_sigmoid
    ipz[s] = ip[((size_t)b * NHD + hd) * SS + s];
  }
  __syncthreads();
  if (s == 0) {
    cs[0] = 0.f;
    for (int i = 0; i < SS; ++i) cs[i + 1] = cs[i] + lfz[i];
  }
  __syncthreads();

  float qr[DHm];
  for (int d = 0; d < DHm; ++d) qr[d] = q[rowbase + d];
  float logd[SS], dots[SS];
  float mx = -1e30f;
  for (int t = 0; t <= s; ++t) {
    float dot = 0.f;
    for (int d = 0; d < DHm; ++d) dot += qr[d] * ksm[t][d];
    dots[t] = dot;
    float ld = cs[s + 1] - cs[t + 1] + ipz[t];
    logd[t] = ld;
    mx = fmaxf(mx, ld);
  }
  const float scale = 0.08838834764831845f;         // 1/sqrt(128)
  float ssum = 0.f;
  for (int t = 0; t <= s; ++t) {
    float c = dots[t] * scale * expf(logd[t] - mx);
    dots[t] = c;
    ssum += c;
  }
  float norm = fmaxf(fabsf(ssum), expf(-mx)) + 1e-6f;
  for (int t = 0; t < SS; ++t) Psm[s][t] = (t <= s) ? dots[t] / norm : 0.f;
  __syncthreads();
  for (int d = 0; d < DHm; ++d) ksm[s][d] = v[rowbase + d];   // stage V
  __syncthreads();
  for (int d = 0; d < DHm; ++d) {
    float acc = 0.f;
    for (int t = 0; t < SS; ++t) acc += Psm[s][t] * ksm[t][d];
    hout[rowbase + d] = acc;                        // safe: q consumed above
  }
}

// --------- mLSTM output gating: (hn + skip*xc) * silu(z) --------------------
__global__ void mlstm_gate_kernel(const float* __restrict__ hn, const float* __restrict__ xc,
                                  const float* __restrict__ up, const float* __restrict__ skip,
                                  float* __restrict__ out, long N) {   // N = TT*INR
  long id = (long)blockIdx.x * blockDim.x + threadIdx.x;
  if (id >= N) return;
  int f = (int)(id % INR);
  long t = id / INR;
  float z = up[(size_t)t * (2 * INR) + INR + f];
  float sz = z / (1.f + expf(-z));
  out[id] = (hn[id] + skip[f] * xc[id]) * sz;
}

// ------------------------- sLSTM sequential scan ----------------------------
__global__ __launch_bounds__(256)
void slstm_scan_kernel(const float* __restrict__ gi, const float* __restrict__ gf,
                       const float* __restrict__ gz, const float* __restrict__ go,
                       const float* __restrict__ ri, const float* __restrict__ rf,
                       const float* __restrict__ rz, const float* __restrict__ ro,
                       float* __restrict__ hs) {
  const int b = blockIdx.x;
  const int tid = threadIdx.x;                 // 256 = NH*DH
  const int n = tid >> 6;                      // head
  const int j = tid & 63;                      // per-head index
  __shared__ float hsm[DM];
  float h = 0.f, c = 0.f, nn = 0.f, m = 0.f;
  hsm[tid] = 0.f;
  __syncthreads();
  const float* Ri = ri + (size_t)(n * 64 + j) * 64;
  const float* Rf = rf + (size_t)(n * 64 + j) * 64;
  const float* Rz = rz + (size_t)(n * 64 + j) * 64;
  const float* Ro = ro + (size_t)(n * 64 + j) * 64;
  for (int s = 0; s < SS; ++s) {
    size_t gidx = ((size_t)(b * SS + s)) * DM + tid;
    float ir = gi[gidx], fr = gf[gidx], zr = gz[gidx], orr = go[gidx];
    const float* hh = hsm + n * 64;
    float di = 0.f, df = 0.f, dz = 0.f, dq = 0.f;
#pragma unroll 8
    for (int jj = 0; jj < 64; ++jj) {
      float hv = hh[jj];
      di += Ri[jj] * hv; df += Rf[jj] * hv; dz += Rz[jj] * hv; dq += Ro[jj] * hv;
    }
    ir += di; fr += df; zr += dz; orr += dq;
    float mn = fmaxf(fr + m, ir);
    float ii = expf(ir - mn), ff = expf(fr + m - mn);
    c = ff * c + ii * tanhf(zr);
    nn = ff * nn + ii;
    h = (1.f / (1.f + expf(-orr))) * c / nn;
    m = mn;
    hs[gidx] = h;
    __syncthreads();
    hsm[tid] = h;
    __syncthreads();
  }
}

// -------------------- FFN gating: gelu(g) * u -------------------------------
__global__ void ffn_gate_kernel(const float* __restrict__ up, float* __restrict__ out, long N) {
  long id = (long)blockIdx.x * blockDim.x + threadIdx.x;
  if (id >= N) return;                              // N = TT*FFD
  int f = (int)(id % FFD);
  long t = id / FFD;
  float g = up[(size_t)t * (2 * FFD) + f];
  float u = up[(size_t)t * (2 * FFD) + FFD + f];
  float ge = 0.5f * g * (1.f + erff(g * 0.70710678118654752f));
  out[(size_t)t * FFD + f] = ge * u;
}

// -------------------- final classifier head ---------------------------------
__global__ void fc_kernel(const float* __restrict__ h, const float* __restrict__ w,
                          const float* __restrict__ b, float* __restrict__ out, int O) {
  int id = blockIdx.x * blockDim.x + threadIdx.x;
  if (id >= BB * O) return;
  int o = id % O, bb = id / O;
  const float* hr = h + ((size_t)(bb * SS + SS - 1)) * DM;
  float acc = b[o];
  for (int d = 0; d < DM; ++d) acc += hr[d] * w[d * O + o];
  out[id] = acc;
}

// ===========================================================================
// Host orchestration
// ===========================================================================
static inline void gemm(hipStream_t st, const float* A, int lda,
                        const float* W, int ldw, int wt,
                        const float* bias, const float* add, int ldadd,
                        float* C, int ldc, int M, int N, int K) {
  int strips = (M / 16) * (N / 64);
  int blocks = (strips + 7) / 8;
  if (wt)
    wmma_gemm_kernel<1><<<blocks, 256, 0, st>>>(A, lda, W, ldw, bias, add, ldadd, C, ldc, M, N, K);
  else
    wmma_gemm_kernel<0><<<blocks, 256, 0, st>>>(A, lda, W, ldw, bias, add, ldadd, C, ldc, M, N, K);
}

struct Ws {
  float *h, *xn, *up, *xc, *qb, *kb, *vb, *ipb, *fpb, *xpad;
};

// mLSTM block: h <- h + block(h)
static void run_mblock(const float* const* P, int m, const Ws& w, hipStream_t st) {
  const long Nfull = (long)TT * INR;
  ln_kernel<<<TT, DM, 0, st>>>(w.h, P[m + 0], w.xn);
  // up projection (256 -> 1024); xm = up[:, :512], z = up[:, 512:]
  gemm(st, w.xn, DM, P[m + 1], 2 * INR, 0, nullptr, nullptr, 0, w.up, 2 * INR, TT, 2 * INR, DM);
  // xc = silu(causal_conv(xm))
  conv_silu_kernel<<<(int)((Nfull + 255) / 256), 256, 0, st>>>(w.up, 2 * INR, P[m + 2], P[m + 3],
                                                              w.xc, INR, INR, Nfull, 1);
  // q,k from xc ; v from xm
  blockdiag_kernel<<<(int)((Nfull + 255) / 256), 256, 0, st>>>(w.xc, INR,     P[m + 4], w.qb, Nfull);
  blockdiag_kernel<<<(int)((Nfull + 255) / 256), 256, 0, st>>>(w.xc, INR,     P[m + 5], w.kb, Nfull);
  blockdiag_kernel<<<(int)((Nfull + 255) / 256), 256, 0, st>>>(w.up, 2 * INR, P[m + 6], w.vb, Nfull);
  // ip/fp gates from concat(q,k,v)
  mlstm_gates_kernel<<<TT, 256, 0, st>>>(w.qb, w.kb, w.vb, P[m + 7], P[m + 8],
                                         P[m + 9], P[m + 10], w.ipb, w.fpb);
  // stabilized parallel cell; output overwrites q buffer
  mlstm_cell_kernel<<<BB * NHD, SS, 0, st>>>(w.qb, w.kb, w.vb, w.ipb, w.fpb, w.qb);
  // per-head norm (groups of 128) -> k buffer
  mhnorm_kernel<<<TT * NHD, 128, 0, st>>>(w.qb, P[m + 11], nullptr, w.kb, INR);
  // (hn + skip*xc) * silu(z) -> v buffer
  mlstm_gate_kernel<<<(int)((Nfull + 255) / 256), 256, 0, st>>>(w.kb, w.xc, w.up, P[m + 12],
                                                               w.vb, Nfull);
  // h = h + gated @ w_down
  gemm(st, w.vb, INR, P[m + 13], DM, 0, nullptr, w.h, DM, w.h, DM, TT, DM, INR);
}

// sLSTM block: h <- h + slstm(h); then h <- h + ffn(ln(h))
static void run_sblock(const float* const* P, int s, const Ws& w, hipStream_t st) {
  const long Nd = (long)TT * DM;
  ln_kernel<<<TT, DM, 0, st>>>(w.h, P[s + 0], w.xn);
  conv_silu_kernel<<<(int)((Nd + 255) / 256), 256, 0, st>>>(w.xn, DM, P[s + 1], P[s + 2],
                                                            w.xc, DM, DM, Nd, 1);
  // gate pre-activations per head via WMMA GEMMs (W stored (o,i) -> WT=1)
  float* gi = w.qb;             float* gf = w.qb + (size_t)TT * DM;
  float* gz = w.kb;             float* go = w.kb + (size_t)TT * DM;
  for (int n = 0; n < NHD; ++n) {
    const int off = n * 64;
    gemm(st, w.xc + off, DM, P[s + 3] + n * 4096, 64, 1, P[s + 7] + off, nullptr, 0,
         gi + off, DM, TT, 64, 64);                            // i-gate (from xc)
    gemm(st, w.xc + off, DM, P[s + 4] + n * 4096, 64, 1, P[s + 10] + off, nullptr, 0,
         gf + off, DM, TT, 64, 64);                            // f-gate (from xc)
    gemm(st, w.xn + off, DM, P[s + 5] + n * 4096, 64, 1, P[s + 8] + off, nullptr, 0,
         gz + off, DM, TT, 64, 64);                            // z-gate (from xn)
    gemm(st, w.xn + off, DM, P[s + 6] + n * 4096, 64, 1, P[s + 9] + off, nullptr, 0,
         go + off, DM, TT, 64, 64);                            // o-gate (from xn)
  }
  // sequential scan -> hs in v buffer
  slstm_scan_kernel<<<BB, DM, 0, st>>>(gi, gf, gz, go,
                                       P[s + 11], P[s + 12], P[s + 13], P[s + 14], w.vb);
  // h = h + mh_norm(hs) (groups of 64)
  mhnorm_kernel<<<TT * NHD, 64, 0, st>>>(w.vb, P[s + 15], w.h, w.h, DM);
  // FFN: h = h + (gelu(g)*u) @ ffn_down
  ln_kernel<<<TT, DM, 0, st>>>(w.h, P[s + 16], w.xn);
  gemm(st, w.xn, DM, P[s + 17], 2 * FFD, 0, nullptr, nullptr, 0, w.up, 2 * FFD, TT, 2 * FFD, DM);
  const long Nf = (long)TT * FFD;
  ffn_gate_kernel<<<(int)((Nf + 255) / 256), 256, 0, st>>>(w.up, w.xc, Nf);
  gemm(st, w.xc, FFD, P[s + 18], DM, 0, nullptr, w.h, DM, w.h, DM, TT, DM, FFD);
}

extern "C" void kernel_launch(void* const* d_in, const int* in_sizes, int n_in,
                              void* d_out, int out_size, void* d_ws, size_t ws_size,
                              hipStream_t stream) {
  (void)in_sizes; (void)n_in; (void)out_size; (void)ws_size;
  const float* P[80];
  for (int i = 0; i < n_in && i < 80; ++i) P[i] = (const float*)d_in[i];

  // Flat param index map (insertion order of setup_inputs()):
  // 0:x 1:w_in 2:b_in | s0: b0 @3..16, b1 @17..35, post_ln @36
  //                   | s1: b0 @37..50, b1 @51..69, post_ln @70 | 71:w_fc 72:b_fc
  const int M0 = 3, S0 = 17, P0 = 36;
  const int M1 = 37, S1 = 51, P1 = 70;

  Ws w;
  float* ws = (float*)d_ws;
  w.h    = ws;
  w.xn   = w.h  + (size_t)TT * DM;
  w.up   = w.xn + (size_t)TT * DM;
  w.xc   = w.up + (size_t)TT * 1024;
  w.qb   = w.xc + (size_t)TT * INR;
  w.kb   = w.qb + (size_t)TT * INR;
  w.vb   = w.kb + (size_t)TT * INR;
  w.ipb  = w.vb + (size_t)TT * INR;
  w.fpb  = w.ipb + (size_t)BB * NHD * SS;
  w.xpad = w.fpb + (size_t)BB * NHD * SS;

  // zero-pad x (K=60 -> 64), then input projection: h = xpad @ w_in + b_in
  const long Np = (long)TT * 64;
  pad_kernel<<<(int)((Np + 255) / 256), 256, 0, stream>>>(P[0], w.xpad, Np);
  gemm(stream, w.xpad, 64, P[1], DM, 0, P[2], nullptr, 0, w.h, DM, TT, DM, 64);

  run_mblock(P, M0, w, stream);
  run_sblock(P, S0, w, stream);
  ln_kernel<<<TT, DM, 0, stream>>>(w.h, P[P0], w.h);

  run_mblock(P, M1, w, stream);
  run_sblock(P, S1, w, stream);
  ln_kernel<<<TT, DM, 0, stream>>>(w.h, P[P1], w.h);

  fc_kernel<<<(BB * 2 + 255) / 256, 256, 0, stream>>>(w.h, P[71], P[72], (float*)d_out, 2);
}
